// SRU_15169824489604
// MI455X (gfx1250) — matrix operand order repaired and checked by reading, not compiled, not run on hardware
//
#include <hip/hip_runtime.h>

// ---------------------------------------------------------------------------
// SRU recurrence on MI455X (gfx1250), bf16 WMMA.
// mu master state (f32) lives in registers (distributed per the WMMA C/D
// layout across the 16 waves); alpha=0 block carries no state (mu == phi).
// LDS holds only bf16 operand shadows; GEMM inner loops are software-
// pipelined ping-pong: loads for iteration i+1 issue before WMMAs of i.
// ---------------------------------------------------------------------------

typedef __attribute__((ext_vector_type(16))) __bf16 v16bf;
typedef __attribute__((ext_vector_type(8)))  __bf16 v8bf;
typedef __attribute__((ext_vector_type(8)))  float  v8f;

#define SEQ   512
#define BATCH 128
#define XDIM  256
#define PHID  512
#define RDIM  256
#define OUTD  256
#define MUD   2048          // PHID * 4 alphas
#define XRD   (XDIM + RDIM) // 512

#define ROWS    16          // batch rows per workgroup (= WMMA M tile)
#define THREADS 512         // 16 waves (wave32)
#define NBLOCKS (BATCH / ROWS)

// GEMM tiling: N tiles of 16, K blocks of 32 (bf16 WMMA 16x16x32)
#define KB1 (MUD / 32)      // 64
#define NT1 (RDIM / 16)     // 16
#define KB2 (XRD / 32)      // 16
#define NT2 (PHID / 16)     // 32
#define KB3 (MUD / 32)      // 64
#define NT3 (OUTD / 16)     // 16

#define NP1 (NT1 * KB1 * 32 * 16)   // 524288 bf16
#define NP2 (NT2 * KB2 * 32 * 16)   // 262144 bf16
#define NP3 (NT3 * KB3 * 32 * 16)   // 524288 bf16

// LDS: muh (16x2048 bf16) + xrh (16x512 bf16) = 80 KB
#define LDS_BYTES ((ROWS * MUD + ROWS * XRD) * 2)

// ---------------------------------------------------------------------------
// Weight packing: per-lane B-fragment layout (ISA 7.12.2, 16-bit B 32x16):
//   lane&15 = N column; h = lane>>4; element i -> K = 8h + (i&7) + 16*(i>>3).
//   16 bf16 per lane, contiguous 32 bytes -> 2x global_load_b128 at use.
// ---------------------------------------------------------------------------
__global__ void sru_pack(const float* __restrict__ W1, const float* __restrict__ W2,
                         const float* __restrict__ W3, __bf16* __restrict__ pW1,
                         __bf16* __restrict__ pW2, __bf16* __restrict__ pW3) {
    int e = blockIdx.x * blockDim.x + threadIdx.x;
    const float* W; __bf16* P; int numKB, Ksrc, rel;
    if (e < NP1)              { W = W1; P = pW1; numKB = KB1; Ksrc = MUD; rel = e; }
    else if (e < NP1 + NP2)   { W = W2; P = pW2; numKB = KB2; Ksrc = XRD; rel = e - NP1; }
    else if (e < NP1+NP2+NP3) { W = W3; P = pW3; numKB = KB3; Ksrc = MUD; rel = e - NP1 - NP2; }
    else return;
    int i    = rel & 15;
    int lane = (rel >> 4) & 31;
    int rest = rel >> 9;               // nTile*numKB + kBlock
    int kB   = rest % numKB;
    int nT   = rest / numKB;
    int n    = nT * 16 + (lane & 15);
    int k    = kB * 32 + ((lane >> 4) << 3) + (i & 7) + ((i >> 3) << 4);
    P[rel] = (__bf16)W[(size_t)n * Ksrc + k];
}

// A-fragment from bf16 row-major LDS: two 16B runs (K+0..7, K+16..23),
// i.e. 2x ds_load_b128, no conversions.
__device__ __forceinline__ v16bf load_a_bf16(const __bf16* __restrict__ base,
                                             int kBlock) {
    const __bf16* p = base + kBlock * 32;
    v8bf lo = *(const v8bf*)(p);
    v8bf hi = *(const v8bf*)(p + 16);
    return __builtin_shufflevector(lo, hi, 0, 1, 2, 3, 4, 5, 6, 7,
                                   8, 9, 10, 11, 12, 13, 14, 15);
}

// Packed B-fragment: 32 contiguous bytes per lane (2x global_load_b128, L2-hot).
// base = pW + (nT*numKB*32 + lane)*16 pre-computed per wave.
__device__ __forceinline__ v16bf load_b(const __bf16* __restrict__ base, int kB) {
    return *(const v16bf*)(base + ((size_t)kB << 9));   // kB*32*16
}

__device__ __forceinline__ v8f wmma_bf16(v16bf a, v16bf b, v8f c) {
    return __builtin_amdgcn_wmma_f32_16x16x32_bf16(false, a, false, b, (short)0, c,
                                                   false, false);
}

// K=2048 GEMM (KB=64): dual accumulator chains, ping-pong double buffering.
// Consumes kb,kb+1 while kb+2,kb+3 are in flight; no fragment copies.
__device__ __forceinline__ v8f gemm_k64(const __bf16* __restrict__ aBase,
                                        const __bf16* __restrict__ bBase) {
    v8f acc0 = {}, acc1 = {};
    v16bf a0 = load_a_bf16(aBase, 0), g0 = load_b(bBase, 0);
    v16bf a1 = load_a_bf16(aBase, 1), g1 = load_b(bBase, 1);
    #pragma unroll 1
    for (int kb = 0; kb < KB1 - 4; kb += 4) {
        v16bf a2 = load_a_bf16(aBase, kb + 2), g2 = load_b(bBase, kb + 2);
        v16bf a3 = load_a_bf16(aBase, kb + 3), g3 = load_b(bBase, kb + 3);
        acc0 = wmma_bf16(a0, g0, acc0);
        acc1 = wmma_bf16(a1, g1, acc1);
        a0 = load_a_bf16(aBase, kb + 4); g0 = load_b(bBase, kb + 4);
        a1 = load_a_bf16(aBase, kb + 5); g1 = load_b(bBase, kb + 5);
        acc0 = wmma_bf16(a2, g2, acc0);
        acc1 = wmma_bf16(a3, g3, acc1);
    }
    {   // epilogue: kb = KB1-4 .. KB1-1
        v16bf a2 = load_a_bf16(aBase, KB1 - 2), g2 = load_b(bBase, KB1 - 2);
        v16bf a3 = load_a_bf16(aBase, KB1 - 1), g3 = load_b(bBase, KB1 - 1);
        acc0 = wmma_bf16(a0, g0, acc0);
        acc1 = wmma_bf16(a1, g1, acc1);
        acc0 = wmma_bf16(a2, g2, acc0);
        acc1 = wmma_bf16(a3, g3, acc1);
    }
    return acc0 + acc1;
}

__global__ __launch_bounds__(THREADS, 1)
void sru_main(const float* __restrict__ inputs,
              const __bf16* __restrict__ pW1, const float* __restrict__ b1,
              const __bf16* __restrict__ pW2, const float* __restrict__ b2,
              const __bf16* __restrict__ pW3, const float* __restrict__ b3,
              float* __restrict__ out) {
    extern __shared__ __align__(16) __bf16 smem[];
    __bf16* muh = smem;              // 16 x 2048 bf16 shadow of mu (GEMM A operand)
    __bf16* xrh = smem + ROWS * MUD; // 16 x 512  bf16 concat [x | r]

    const int tid     = threadIdx.x;
    const int lane    = tid & 31;
    const int w       = tid >> 5;              // wave id 0..15
    const int rowBase = blockIdx.x * ROWS;
    const int mBase   = (lane >> 4) << 3;      // accumulator row base (C/D layout)
    const int nCol    = lane & 15;             // accumulator column (C/D layout)
    // carried alphas (alpha=0 block needs no state: mu == phi there)
    const float ALPHA[3] = {0.5f, 0.9f, 0.99f};

    // Per-wave loop-invariant operand bases.
    const __bf16* muhA = muh + (lane & 15) * MUD + ((lane >> 4) << 3);
    const __bf16* xrhA = xrh + (lane & 15) * XRD + ((lane >> 4) << 3);
    const __bf16* bW1  = pW1 + (((size_t)w * KB1) * 32 + lane) * 16;
    const __bf16* bW2a = pW2 + (((size_t)(2 * w) * KB2) * 32 + lane) * 16;
    const __bf16* bW2b = pW2 + (((size_t)(2 * w + 1) * KB2) * 32 + lane) * 16;
    const __bf16* bW3  = pW3 + (((size_t)w * KB3) * 32 + lane) * 16;

    // Loop-invariant biases (L2-hot, load once).
    const float bias1  = b1[w * 16 + nCol];
    const float bias2a = b2[(2 * w) * 16 + nCol];
    const float bias2b = b2[(2 * w + 1) * 16 + nCol];

    // f32 master mu state in registers for alpha in {0.5, 0.9, 0.99}:
    // wave w owns phi-tiles 2w, 2w+1. Layout == WMMA C/D accumulator layout.
    v8f mureg[2][3];
    #pragma unroll
    for (int tt = 0; tt < 2; ++tt)
        #pragma unroll
        for (int al = 0; al < 3; ++al)
            mureg[tt][al] = (v8f){};

    // zero the bf16 shadow (mu0 = 0)
    {
        unsigned int* z = (unsigned int*)muh;
        for (int i = tid; i < (ROWS * MUD) / 2; i += THREADS) z[i] = 0u;
    }
    __syncthreads();

    const int xrow = tid >> 5;        // 0..15
    const int xcol = (tid & 31) * 8;  // 0..248

    for (int t = 0; t < SEQ; ++t) {
        // ---- Phase A: stage x_t (bf16) into xrh[:, 0:256]; prefetch x_{t+1}.
        // Disjoint LDS from Phase B and B never reads x -> no A/B barrier.
        {
            const float* src = inputs + (((size_t)t * BATCH + rowBase + xrow) * XDIM + xcol);
            float4 f0 = *(const float4*)src;
            float4 f1 = *(const float4*)(src + 4);
            v8bf xv;
            xv[0] = (__bf16)f0.x; xv[1] = (__bf16)f0.y;
            xv[2] = (__bf16)f0.z; xv[3] = (__bf16)f0.w;
            xv[4] = (__bf16)f1.x; xv[5] = (__bf16)f1.y;
            xv[6] = (__bf16)f1.z; xv[7] = (__bf16)f1.w;
            *(v8bf*)&xrh[xrow * XRD + xcol] = xv;
            if (t + 1 < SEQ)
                __builtin_prefetch(inputs + (((size_t)(t + 1) * BATCH + rowBase + xrow) * XDIM + xcol), 0, 1);
        }

        // ---- Phase B: r = relu(muh @ W1^T + b1) -> xrh[:, 256:512] ---------
        // 16 N-tiles, one per wave; K = 2048, pipelined dual-chain WMMA.
        {
            v8f acc = gemm_k64(muhA, bW1);
            int col = XDIM + w * 16 + nCol;
            #pragma unroll
            for (int v = 0; v < 8; ++v)
                xrh[(mBase + v) * XRD + col] = (__bf16)fmaxf(acc[v] + bias1, 0.0f);
        }
        __syncthreads();

        // ---- Phase C: phi = relu(xrh @ W2^T + b2);
        //      mu = a*mu + (1-a)*phi (registers), refresh bf16 shadow --------
        // 32 N-tiles, two per wave sharing the A fragment; K = 512, pipelined.
        {
            v8f acc0 = {}, acc1 = {};
            v16bf a = load_a_bf16(xrhA, 0);
            v16bf h0 = load_b(bW2a, 0), h1 = load_b(bW2b, 0);
            #pragma unroll 1
            for (int kb = 0; kb < KB2 - 2; kb += 2) {
                v16bf an  = load_a_bf16(xrhA, kb + 1);
                v16bf h0n = load_b(bW2a, kb + 1), h1n = load_b(bW2b, kb + 1);
                acc0 = wmma_bf16(a, h0, acc0);
                acc1 = wmma_bf16(a, h1, acc1);
                a  = load_a_bf16(xrhA, kb + 2);
                h0 = load_b(bW2a, kb + 2); h1 = load_b(bW2b, kb + 2);
                acc0 = wmma_bf16(an, h0n, acc0);
                acc1 = wmma_bf16(an, h1n, acc1);
            }
            {   // epilogue: kb = KB2-2, KB2-1
                v16bf an  = load_a_bf16(xrhA, KB2 - 1);
                v16bf h0n = load_b(bW2a, KB2 - 1), h1n = load_b(bW2b, KB2 - 1);
                acc0 = wmma_bf16(a, h0, acc0);
                acc1 = wmma_bf16(a, h1, acc1);
                acc0 = wmma_bf16(an, h0n, acc0);
                acc1 = wmma_bf16(an, h1n, acc1);
            }
            #pragma unroll
            for (int tt = 0; tt < 2; ++tt) {
                int col0 = (2 * w + tt) * 16 + nCol;
                float bias = tt ? bias2b : bias2a;
                v8f acc = tt ? acc1 : acc0;
                #pragma unroll
                for (int v = 0; v < 8; ++v) {
                    float ph  = fmaxf(acc[v] + bias, 0.0f);
                    int   rb  = (mBase + v) * MUD + col0;
                    // alpha = 0: mu == phi, no carried state
                    muh[rb] = (__bf16)ph;
                    #pragma unroll
                    for (int al = 0; al < 3; ++al) {
                        float m = ALPHA[al] * mureg[tt][al][v] + (1.0f - ALPHA[al]) * ph;
                        mureg[tt][al][v] = m;
                        muh[rb + (al + 1) * PHID] = (__bf16)m;
                    }
                }
            }
        }
        __syncthreads();
    }

    // ---- Final: out = relu(muh @ W3^T + b3) --------------------------------
    {
        v8f acc = gemm_k64(muhA, bW3);
        float bias = b3[w * 16 + nCol];
        #pragma unroll
        for (int v = 0; v < 8; ++v) {
            int grow = rowBase + mBase + v;
            out[(size_t)grow * OUTD + w * 16 + nCol] = fmaxf(acc[v] + bias, 0.0f);
        }
    }
}

extern "C" void kernel_launch(void* const* d_in, const int* in_sizes, int n_in,
                              void* d_out, int out_size, void* d_ws, size_t ws_size,
                              hipStream_t stream) {
    const float* inputs = (const float*)d_in[0];
    const float* W1     = (const float*)d_in[1];  // W_mu2r   (256, 2048)
    const float* b1     = (const float*)d_in[2];
    const float* W2     = (const float*)d_in[3];  // W_xr2phi (512, 512)
    const float* b2     = (const float*)d_in[4];
    const float* W3     = (const float*)d_in[5];  // W_mu2o   (256, 2048)
    const float* b3     = (const float*)d_in[6];

    __bf16* pW1 = (__bf16*)d_ws;
    __bf16* pW2 = (__bf16*)((char*)d_ws + (size_t)NP1 * 2);
    __bf16* pW3 = (__bf16*)((char*)d_ws + ((size_t)NP1 + NP2) * 2);

    int total = NP1 + NP2 + NP3;
    sru_pack<<<(total + 255) / 256, 256, 0, stream>>>(W1, W2, W3, pW1, pW2, pW3);

    (void)hipFuncSetAttribute((const void*)sru_main,
                              hipFuncAttributeMaxDynamicSharedMemorySize, LDS_BYTES);
    sru_main<<<NBLOCKS, THREADS, LDS_BYTES, stream>>>(inputs, pW1, b1, pW2, b2,
                                                      pW3, b3, (float*)d_out);
}
